// enc_mtan_rnn_240518168904
// MI455X (gfx1250) — matrix operand order repaired and checked by reading, not compiled, not run on hardware
//
#include <hip/hip_runtime.h>
#include <math.h>

typedef __attribute__((ext_vector_type(16))) _Float16 v16h;
typedef __attribute__((ext_vector_type(8)))  float    v8f;

#define BD   8
#define TKD  512
#define TQD  128
#define ED   128
#define HD   128
#define DDIM 82   // 2*INPUT_DIM
#define IDIM 41

// ---------------- WMMA tile loaders (contiguous k order, same for A and B) ----------------
// A tile 16x32 f16: lane L -> row m0+(L&15); element i -> K = k0 + (L>>4)*16 + i
template<int KK, int LDA>
__device__ __forceinline__ v16h load_a(const float* __restrict__ A, int m0, int k0)
{
  const int lane = threadIdx.x & 31;
  const int half = lane >> 4;
  const float* __restrict__ arow = A + (long long)(m0 + (lane & 15)) * LDA;
  v16h a;
  if (k0 + 32 <= KK) {                 // constexpr after full unroll of the K loop
#pragma unroll
    for (int i = 0; i < 16; ++i)
      a[i] = (_Float16)arow[k0 + half * 16 + i];
  } else {                             // K tail (only for KK % 32 != 0)
#pragma unroll
    for (int i = 0; i < 16; ++i) {
      int kk = k0 + half * 16 + i;
      float v = (kk < KK) ? arow[kk] : 0.0f;
      a[i] = (_Float16)v;
    }
  }
  return a;
}

// B tile 32x16 f16 (K x N): lane L -> col n0+(L&15); element i -> K = k0 + (L>>4)*16 + i
template<int KK, int LDB, bool TRANSB, bool CEDGE, int NN>
__device__ __forceinline__ v16h load_b(const float* __restrict__ Bm, int n0, int k0)
{
  const int lane = threadIdx.x & 31;
  const int col  = n0 + (lane & 15);
  const int kb   = k0 + ((lane >> 4) << 4);
  v16h b;
  if (!CEDGE && (k0 + 32 <= KK)) {     // constexpr: interior tile, unconditional loads
    if (TRANSB) {
      const float* __restrict__ brow = Bm + (long long)col * LDB + kb;
#pragma unroll
      for (int i = 0; i < 16; ++i) b[i] = (_Float16)brow[i];
    } else {
      const float* __restrict__ bp = Bm + (long long)kb * LDB + col;
#pragma unroll
      for (int i = 0; i < 16; ++i) b[i] = (_Float16)bp[(long long)i * LDB];
    }
  } else {
    const bool cok = !CEDGE || (col < NN);
#pragma unroll
    for (int i = 0; i < 16; ++i) {
      int kk = kb + i;
      float v = 0.0f;
      if (cok && kk < KK)
        v = TRANSB ? Bm[(long long)col * LDB + kk] : Bm[(long long)kk * LDB + col];
      b[i] = (_Float16)v;
    }
  }
  return b;
}

// C store: lane L -> col n0+(L&15); VGPR j -> row m0 + j (+8 for lanes 16..31)
template<int NN, int LDC, bool CEDGE>
__device__ __forceinline__ void store_c(float* __restrict__ C,
                                        const float* __restrict__ bias, v8f acc,
                                        int m0, int n0, float scale, int doRelu)
{
  const int lane = threadIdx.x & 31;
  const int col  = n0 + (lane & 15);
  const int half = lane >> 4;
  if (!CEDGE || col < NN) {
    float bv = bias ? bias[col] : 0.0f;
#pragma unroll
    for (int j = 0; j < 8; ++j) {
      float v = acc[j] * scale + bv;
      if (doRelu) v = fmaxf(v, 0.0f);
      C[(long long)(m0 + j + half * 8) * LDC + col] = v;
    }
  }
}

// ------- Fully-specialized batched WMMA GEMM: C = scale*(A x B) + bias, opt. relu -----
// Each wave computes a 16x32 strip (two 16x16 N-tiles), reusing one A tile per K-step.
// TRANSB: B element (k,n) read from Bm[n*LDB + k]  (W row-major (N,K): computes A @ W^T)
template<int MM, int NN, int KK, int LDA, int LDB, int LDC, bool TRANSB>
__global__ void __launch_bounds__(256)
wmma_gemm_t(const float* __restrict__ A, const float* __restrict__ Bm,
            const float* __restrict__ bias, float* __restrict__ C,
            long long sA, long long sB, long long sC, float scale, int doRelu)
{
  static_assert(MM % 16 == 0, "M must be a multiple of 16");
  constexpr int NT    = (NN + 31) / 32;
  constexpr int MT    = MM / 16;
  constexpr int TOTAL = NT * MT;
  const int wave = blockIdx.x * 8 + (threadIdx.x >> 5);
  if (wave >= TOTAL) return;                 // wave-uniform: EXEC stays all-1s for WMMA
  const int m0 = (wave / NT) * 16;
  const int n0 = (wave % NT) * 32;
  const int n1 = n0 + 16;
  const long long bz = blockIdx.y;
  A += sA * bz; Bm += sB * bz; C += sC * bz;

  constexpr bool N16 = (NN % 16 == 0);
  constexpr bool N32 = (NN % 32 == 0);
  const bool full0 = N16 || (n0 + 16 <= NN); // folds to true for aligned N
  const bool do1   = N32 || (n1 < NN);
  const bool full1 = N16 || (n1 + 16 <= NN);

  v8f acc0 = {}, acc1 = {};
#pragma unroll
  for (int k0 = 0; k0 < KK; k0 += 32) {
    v16h a  = load_a<KK, LDA>(A, m0, k0);
    v16h b0 = full0 ? load_b<KK, LDB, TRANSB, false, NN>(Bm, n0, k0)
                    : load_b<KK, LDB, TRANSB, true,  NN>(Bm, n0, k0);
    acc0 = __builtin_amdgcn_wmma_f32_16x16x32_f16(false, a, false, b0, (short)0, acc0, false, false);
    if (do1) {
      v16h b1 = full1 ? load_b<KK, LDB, TRANSB, false, NN>(Bm, n1, k0)
                      : load_b<KK, LDB, TRANSB, true,  NN>(Bm, n1, k0);
      acc1 = __builtin_amdgcn_wmma_f32_16x16x32_f16(false, a, false, b1, (short)0, acc1, false, false);
    }
  }
  if (full0) store_c<NN, LDC, false>(C, bias, acc0, m0, n0, scale, doRelu);
  else       store_c<NN, LDC, true >(C, bias, acc0, m0, n0, scale, doRelu);
  if (do1) {
    if (full1) store_c<NN, LDC, false>(C, bias, acc1, m0, n1, scale, doRelu);
    else       store_c<NN, LDC, true >(C, bias, acc1, m0, n1, scale, doRelu);
  }
}

template<int MM, int NN, int KK, int LDA, int LDB, int LDC, bool TRANSB>
static void launch_gemm(const float* A, const float* Bm, const float* bias, float* C,
                        long long sA, long long sB, long long sC,
                        float scale, int relu, int batch, hipStream_t stream)
{
  constexpr int TOTAL = (MM / 16) * ((NN + 31) / 32);
  dim3 grid((TOTAL + 7) / 8, batch);
  wmma_gemm_t<MM, NN, KK, LDA, LDB, LDC, TRANSB>
      <<<grid, 256, 0, stream>>>(A, Bm, bias, C, sA, sB, sC, scale, relu);
}

// ---------------- time embeddings (linear channel 0, sin channels 1..127) --------------
__global__ void time_embed_kernel(const float* __restrict__ ts, const float* __restrict__ query,
                                  const float* __restrict__ e1_wp, const float* __restrict__ e1_bp,
                                  const float* __restrict__ e1_wl, const float* __restrict__ e1_bl,
                                  const float* __restrict__ e2_wp, const float* __restrict__ e2_bp,
                                  const float* __restrict__ e2_wl, const float* __restrict__ e2_bl,
                                  float* __restrict__ k_emb, float* __restrict__ q_emb)
{
  int idx = blockIdx.x * blockDim.x + threadIdx.x;
  const int NK = BD * TKD * ED;
  if (idx < NK) {
    int r = idx >> 7, c = idx & 127;
    float t = ts[r];
    float v = (c == 0) ? (t * e1_wl[0] + e1_bl[0]) : __sinf(t * e1_wp[c - 1] + e1_bp[c - 1]);
    k_emb[idx] = v;
  } else if (idx < NK + TQD * ED) {
    int j = idx - NK;
    int r = j >> 7, c = j & 127;
    float t = query[r];
    float v = (c == 0) ? (t * e2_wl[0] + e2_bl[0]) : __sinf(t * e2_wp[c - 1] + e2_bp[c - 1]);
    q_emb[j] = v;
  }
}

// ---------------- per-row max + exp (one wave32 per row of 512 scores) -----------------
__global__ void rowmax_exp_kernel(float* __restrict__ S)
{
  int wave = blockIdx.x * 8 + (threadIdx.x >> 5);
  if (wave >= BD * TQD) return;
  int lane = threadIdx.x & 31;
  float* row = S + (long long)wave * TKD;
  float m = -1e30f;
#pragma unroll
  for (int c = 0; c < TKD / 32; ++c) m = fmaxf(m, row[lane + c * 32]);
#pragma unroll
  for (int off = 16; off > 0; off >>= 1) m = fmaxf(m, __shfl_xor(m, off, 32));
#pragma unroll
  for (int c = 0; c < TKD / 32; ++c) {
    int idx = lane + c * 32;
    row[idx] = __expf(row[idx] - m);
  }
}

// ---------------- att finalize: att[:, :41] = numer/denom, att[:, 41:] = (denom>0) -----
__global__ void att_finalize_kernel(const float* __restrict__ numer, float* __restrict__ att)
{
  int idx = blockIdx.x * blockDim.x + threadIdx.x;
  if (idx >= BD * TQD * DDIM) return;
  int m = idx / DDIM, d = idx - m * DDIM;
  const float* nr = numer + (long long)m * DDIM;
  float denom = nr[(d < IDIM) ? (IDIM + d) : d];
  float v;
  if (d < IDIM) v = (denom > 0.0f) ? (nr[d] / denom) : 0.0f;
  else          v = (denom > 0.0f) ? 1.0f : 0.0f;
  att[idx] = v;
}

// ---------------- bidirectional GRU: one persistent block per direction ----------------
__global__ void __launch_bounds__(1024)
gru_kernel(const float* __restrict__ gi_f, const float* __restrict__ gi_b,
           const float* __restrict__ gwh_f, const float* __restrict__ gbh_f,
           const float* __restrict__ gwh_b, const float* __restrict__ gbh_b,
           float* __restrict__ hcat)
{
  const int dir = blockIdx.x;
  const float* gi = dir ? gi_b  : gi_f;
  const float* Wh = dir ? gwh_b : gwh_f;
  const float* bh = dir ? gbh_b : gbh_f;

  __shared__ float h[BD][HD];        // 4 KB
  __shared__ float gh[BD][3 * HD];   // 12 KB

  const int tid = threadIdx.x;       // 1024 threads == BD*HD
  h[tid >> 7][tid & 127] = 0.0f;
  __syncthreads();

  for (int s = 0; s < TQD; ++s) {
    const int t = dir ? (TQD - 1 - s) : s;
    // gh = h @ Wh^T + bh : 8*384 outputs, 3 per thread
#pragma unroll
    for (int rep = 0; rep < 3; ++rep) {
      int w = tid + rep * 1024;
      int b = w / (3 * HD), n = w - b * (3 * HD);
      const float* wrow = Wh + (long long)n * HD;
      float acc = bh[n];
#pragma unroll 4
      for (int k = 0; k < HD; ++k) acc += h[b][k] * wrow[k];
      gh[b][n] = acc;
    }
    __syncthreads();
    // gate combine + state update: one (b, j) cell per thread
    {
      const int b = tid >> 7, j = tid & 127;
      const float* gir = gi + ((long long)b * TQD + t) * (3 * HD);
      float ir = gir[j], iz = gir[HD + j], in = gir[2 * HD + j];
      float hr = gh[b][j], hz = gh[b][HD + j], hn = gh[b][2 * HD + j];
      float r = 1.0f / (1.0f + __expf(-(ir + hr)));
      float z = 1.0f / (1.0f + __expf(-(iz + hz)));
      float n = tanhf(in + r * hn);
      float hnew = (1.0f - z) * n + z * h[b][j];
      h[b][j] = hnew;
      hcat[((long long)b * TQD + t) * (2 * HD) + dir * HD + j] = hnew;
    }
    __syncthreads();
  }
}

// ======================================================================================
extern "C" void kernel_launch(void* const* d_in, const int* in_sizes, int n_in,
                              void* d_out, int out_size, void* d_ws, size_t ws_size,
                              hipStream_t stream)
{
  (void)in_sizes; (void)n_in; (void)out_size; (void)ws_size;
  const float* x      = (const float*)d_in[0];
  const float* ts     = (const float*)d_in[1];
  const float* query  = (const float*)d_in[2];
  const float* e1_wp  = (const float*)d_in[3];
  const float* e1_bp  = (const float*)d_in[4];
  const float* e1_wl  = (const float*)d_in[5];
  const float* e1_bl  = (const float*)d_in[6];
  const float* e2_wp  = (const float*)d_in[7];
  const float* e2_bp  = (const float*)d_in[8];
  const float* e2_wl  = (const float*)d_in[9];
  const float* e2_bl  = (const float*)d_in[10];
  const float* Wq     = (const float*)d_in[11];
  const float* bq     = (const float*)d_in[12];
  const float* Wk     = (const float*)d_in[13];
  const float* bk     = (const float*)d_in[14];
  const float* Wo     = (const float*)d_in[15];
  const float* bo     = (const float*)d_in[16];
  const float* gwi_f  = (const float*)d_in[17];
  const float* gwh_f  = (const float*)d_in[18];
  const float* gbi_f  = (const float*)d_in[19];
  const float* gbh_f  = (const float*)d_in[20];
  const float* gwi_b  = (const float*)d_in[21];
  const float* gwh_b  = (const float*)d_in[22];
  const float* gbi_b  = (const float*)d_in[23];
  const float* gbh_b  = (const float*)d_in[24];
  const float* W1     = (const float*)d_in[25];
  const float* b1     = (const float*)d_in[26];
  const float* W2     = (const float*)d_in[27];
  const float* b2     = (const float*)d_in[28];
  float* out = (float*)d_out;

  // ---- workspace layout (floats) ----
  float* ws    = (float*)d_ws;
  float* k_emb = ws;                         // 4096*128
  float* q_emb = k_emb + 4096 * 128;         // 128*128
  float* Kmat  = q_emb + 128 * 128;          // 4096*128
  float* Qmat  = Kmat  + 4096 * 128;         // 128*128
  float* S     = Qmat  + 128 * 128;          // 8*128*512 (exp in place)
  float* numer = S     + 8 * 128 * 512;      // 8*128*82
  float* att   = numer + 1024 * 82;          // 1024*82
  float* outWo = att   + 1024 * 82;          // 1024*128
  float* gif   = outWo + 1024 * 128;         // 1024*384
  float* gib   = gif   + 1024 * 384;         // 1024*384
  float* hcat  = gib   + 1024 * 384;         // 1024*256
  float* hid   = hcat  + 1024 * 256;         // 1024*50

  // 1) time embeddings
  {
    int total = (BD * TKD + TQD) * ED;
    time_embed_kernel<<<(total + 255) / 256, 256, 0, stream>>>(
        ts, query, e1_wp, e1_bp, e1_wl, e1_bl, e2_wp, e2_bp, e2_wl, e2_bl, k_emb, q_emb);
  }
  // 2) K = k_emb @ Wk^T + bk ; Q = q_emb @ Wq^T + bq
  launch_gemm<BD * TKD, ED, ED, ED, ED, ED, true>(k_emb, Wk, bk, Kmat, 0, 0, 0, 1.0f, 0, 1, stream);
  launch_gemm<TQD,      ED, ED, ED, ED, ED, true>(q_emb, Wq, bq, Qmat, 0, 0, 0, 1.0f, 0, 1, stream);
  // 3) scores: S_b = Q @ K_b^T / sqrt(128)   (A shared across batch -> strideA = 0)
  launch_gemm<TQD, TKD, ED, ED, ED, TKD, true>(Qmat, Kmat, nullptr, S,
      0, (long long)TKD * ED, (long long)TQD * TKD, 0.088388347648318447f, 0, BD, stream);
  // 4) row softmax numerators: E = exp(S - rowmax)
  rowmax_exp_kernel<<<(BD * TQD) / 8, 256, 0, stream>>>(S);
  // 5) numer_b = E_b @ x_b   (x already = vals*obs || obs, so p·x collapses to this)
  launch_gemm<TQD, DDIM, TKD, TKD, DDIM, DDIM, false>(S, x, nullptr, numer,
      (long long)TQD * TKD, (long long)TKD * DDIM, (long long)TQD * DDIM, 1.0f, 0, BD, stream);
  // 6) att = numer / denom  (denom = numer[:, 41:82])
  att_finalize_kernel<<<(BD * TQD * DDIM + 255) / 256, 256, 0, stream>>>(numer, att);
  // 7) out = att @ Wo^T + bo
  launch_gemm<BD * TQD, HD, DDIM, DDIM, DDIM, HD, true>(att, Wo, bo, outWo, 0, 0, 0, 1.0f, 0, 1, stream);
  // 8) GRU input projections (time-independent part, hoisted out of the scan)
  launch_gemm<BD * TQD, 3 * HD, HD, HD, HD, 3 * HD, true>(outWo, gwi_f, gbi_f, gif, 0, 0, 0, 1.0f, 0, 1, stream);
  launch_gemm<BD * TQD, 3 * HD, HD, HD, HD, 3 * HD, true>(outWo, gwi_b, gbi_b, gib, 0, 0, 0, 1.0f, 0, 1, stream);
  // 9) sequential bidirectional GRU (both directions concurrently)
  gru_kernel<<<2, 1024, 0, stream>>>(gif, gib, gwh_f, gbh_f, gwh_b, gbh_b, hcat);
  // 10) head: relu(hcat @ W1^T + b1) @ W2^T + b2 -> out
  launch_gemm<BD * TQD, 50, 2 * HD, 2 * HD, 2 * HD, 50, true>(hcat, W1, b1, hid, 0, 0, 0, 1.0f, 1, 1, stream);
  launch_gemm<BD * TQD, 40, 50, 50, 50, 40, true>(hid, W2, b2, out, 0, 0, 0, 1.0f, 0, 1, stream);
}
